// CTLSTM_62792421867819
// MI455X (gfx1250) — matrix operand order repaired and checked by reading, not compiled, not run on hardware
//
#include <hip/hip_runtime.h>

// CTLSTM (Neural Hawkes) for MI455X / gfx1250.
// Per-step fused GEMM G=[x|h]@[Wx;Wh]+b via v_wmma_f32_16x16x32_bf16,
// weights pre-swizzled into B-fragment layout (L2-resident, 5.5MB bf16).

#define B_   64
#define T_   200
#define D_   256
#define H_   512
#define G7   3584            // 7*H
#define K_   768             // D+H
#define KSTEPS 24            // K_/32
#define NT   224             // G7/16
#define SB   ((size_t)B_*T_*H_)   // stride between stacked outputs (c,cbar,delta,o)

typedef __attribute__((ext_vector_type(16))) __bf16 v16bf;
typedef __attribute__((ext_vector_type(8)))  __bf16 v8bf;
typedef __attribute__((ext_vector_type(8)))  float  v8f;

__device__ __forceinline__ float sigm_(float x)  { return 1.0f / (1.0f + __expf(-x)); }
__device__ __forceinline__ float softplus_(float x) {
    return fmaxf(x, 0.0f) + log1pf(__expf(-fabsf(x)));
}

// ---------------------------------------------------------------------------
// Pre-swizzle W = [Wx; Wh] (768 x 3584 f32) into bf16 B-fragment blocks.
// Block (nt, ks) = 16 cols x 32 K, 512 halfs, laid out so lane L of a wave
// reads its v16bf fragment as one contiguous 32B chunk at lane*32:
//   n = nt*16 + (L&15),  K = ks*32 + j + ((L>=16)?16:0), j = half index 0..15
// ---------------------------------------------------------------------------
__global__ void prep_w(const float* __restrict__ Wx, const float* __restrict__ Wh,
                       __bf16* __restrict__ Wswz) {
    int idx = blockIdx.x * blockDim.x + threadIdx.x;
    if (idx >= NT * KSTEPS * 512) return;
    int within = idx & 511;
    int block  = idx >> 9;
    int ks   = block % KSTEPS;
    int nt   = block / KSTEPS;
    int lane = within >> 4;
    int j    = within & 15;
    int n = nt * 16 + (lane & 15);
    int k = ks * 32 + j + ((lane >> 4) << 4);
    float w = (k < D_) ? Wx[(size_t)k * G7 + n] : Wh[(size_t)(k - D_) * G7 + n];
    Wswz[idx] = (__bf16)w;
}

// input_ (B,T,D) f32 -> bf16, row-major (contiguous K for A staging)
__global__ void prep_x(const float* __restrict__ x, __bf16* __restrict__ xbf) {
    int idx = blockIdx.x * blockDim.x + threadIdx.x;
    if (idx >= B_ * T_ * D_) return;
    xbf[idx] = (__bf16)x[idx];
}

// ---------------------------------------------------------------------------
// One recurrence step. Grid: 8 WGs (h-slice of 64 each), 256 threads (8 waves).
// Wave w: M-tile mi=w&3 (16 rows of batch), r in {rbase,rbase+1} (16-col slices),
// all 7 gate groups -> 14 f32 accumulators; elementwise update stays in VGPRs.
// ---------------------------------------------------------------------------
__global__ __launch_bounds__(256) void ctlstm_step(
    const __bf16* __restrict__ Wswz,
    const __bf16* __restrict__ xbf,
    const float*  __restrict__ dur,
    const float*  __restrict__ bias,
    float* out, int t)
{
    __shared__ __bf16 Alds[64][136];   // 128-K chunk of A=[x|h], +8 halfs pad

    const int j     = blockIdx.x;        // h-slice 0..7 -> h0 = 64*j
    const int tid   = threadIdx.x;
    const int wave  = tid >> 5;
    const int lane  = tid & 31;
    const int mi    = wave & 3;
    const int rbase = (wave >> 2) << 1;  // {0,1} or {2,3}
    const int lhi   = lane >> 4;
    const int l15   = lane & 15;

    v8f acc[14];
    #pragma unroll
    for (int i = 0; i < 14; ++i) acc[i] = {};

    const int am = tid >> 2;            // A-fill: row 0..63
    const int aq = (tid & 3) * 32;      // col base within 128-chunk

    for (int kc = 0; kc < K_; kc += 128) {
        // ---- stage A chunk into LDS ----
        if (kc + 128 <= D_) {                      // x-only chunk
            const __bf16* src = xbf + ((size_t)(am * T_ + t) * D_ + kc + aq);
            #pragma unroll
            for (int u = 0; u < 4; ++u)
                *(uint4*)&Alds[am][aq + u * 8] = *(const uint4*)(src + u * 8);
        } else {                                   // h chunk (decayed hidden)
            const int hh0 = kc - D_ + aq;
            if (t == 0) {
                #pragma unroll
                for (int u = 0; u < 4; ++u)
                    *(uint4*)&Alds[am][aq + u * 8] = make_uint4(0u, 0u, 0u, 0u);
            } else {
                const float dtb = dur[am * T_ + t];
                const size_t off = (size_t)(am * T_ + (t - 1)) * H_ + hh0;
                #pragma unroll 8
                for (int u = 0; u < 32; ++u) {
                    float cp = out[0 * SB + off + u];
                    float cb = out[1 * SB + off + u];
                    float de = out[2 * SB + off + u];
                    float oo = out[3 * SB + off + u];
                    float ct = cb + (cp - cb) * __expf(-de * dtb);
                    Alds[am][aq + u] = (__bf16)(oo * tanhf(ct));
                }
            }
        }
        __syncthreads();

        // ---- WMMA over this K chunk (4 k-steps of 32) ----
        const int ks0 = kc >> 5;
        #pragma unroll
        for (int ksl = 0; ksl < 4; ++ksl) {
            const int mrow = mi * 16 + l15;
            const int kA   = ksl * 32 + (lhi ? 16 : 0);
            v8bf alo = *(const v8bf*)&Alds[mrow][kA];
            v8bf ahi = *(const v8bf*)&Alds[mrow][kA + 8];
            v16bf aF = __builtin_shufflevector(alo, ahi,
                0,1,2,3,4,5,6,7,8,9,10,11,12,13,14,15);
            #pragma unroll
            for (int g = 0; g < 7; ++g) {
                #pragma unroll
                for (int rr = 0; rr < 2; ++rr) {
                    const int ntile = g * 32 + j * 4 + rbase + rr;
                    const __bf16* bp = Wswz
                        + (((size_t)ntile * KSTEPS + (ks0 + ksl)) << 9)
                        + lane * 16;
                    v16bf bF = *(const v16bf*)bp;   // 32B, L2-resident
                    acc[g * 2 + rr] = __builtin_amdgcn_wmma_f32_16x16x32_bf16(
                        false, aF, false, bF, (short)0, acc[g * 2 + rr],
                        false, false);
                }
            }
        }
        __syncthreads();
    }

    // ---- elementwise cell update, all gates live in registers ----
    #pragma unroll
    for (int rr = 0; rr < 2; ++rr) {
        const int r = rbase + rr;
        const int h = j * 64 + r * 16 + l15;
        float bi[7];
        #pragma unroll
        for (int g = 0; g < 7; ++g) bi[g] = bias[g * H_ + h];
        #pragma unroll
        for (int v = 0; v < 8; ++v) {
            const int brow = mi * 16 + lhi * 8 + v;   // C/D layout: M = v + 8*lhi
            float gi  = acc[0 * 2 + rr][v] + bi[0];
            float gf  = acc[1 * 2 + rr][v] + bi[1];
            float gz  = acc[2 * 2 + rr][v] + bi[2];
            float go  = acc[3 * 2 + rr][v] + bi[3];
            float gib = acc[4 * 2 + rr][v] + bi[4];
            float gfb = acc[5 * 2 + rr][v] + bi[5];
            float gd  = acc[6 * 2 + rr][v] + bi[6];
            float ii = sigm_(gi), ff = sigm_(gf), zz = tanhf(gz), on = sigm_(go);
            float ib = sigm_(gib), fb = sigm_(gfb);
            float dn = softplus_(gd);
            float ct = 0.0f, cb = 0.0f;
            if (t > 0) {
                const size_t offp = (size_t)(brow * T_ + (t - 1)) * H_ + h;
                float cp  = out[offp];
                cb        = out[SB + offp];
                float de  = out[2 * SB + offp];
                float dtb = dur[brow * T_ + t];
                ct = cb + (cp - cb) * __expf(-de * dtb);
            }
            float cn  = ff * ct + ii * zz;
            float cbn = fb * cb + ib * zz;
            const size_t offc = (size_t)(brow * T_ + t) * H_ + h;
            out[0 * SB + offc] = cn;
            out[1 * SB + offc] = cbn;
            out[2 * SB + offc] = dn;
            out[3 * SB + offc] = on;
        }
    }
}

extern "C" void kernel_launch(void* const* d_in, const int* in_sizes, int n_in,
                              void* d_out, int out_size, void* d_ws, size_t ws_size,
                              hipStream_t stream) {
    (void)in_sizes; (void)n_in; (void)out_size; (void)ws_size;
    const float* x    = (const float*)d_in[0];   // (B,T,D)
    const float* dur  = (const float*)d_in[1];   // (B,T)
    const float* Wx   = (const float*)d_in[2];   // (D,7H)
    const float* Wh   = (const float*)d_in[3];   // (H,7H)
    const float* bias = (const float*)d_in[4];   // (7H,)
    float* out = (float*)d_out;                  // (4,B,T,H)

    __bf16* Wswz = (__bf16*)d_ws;                                   // 5.25 MB
    __bf16* xbf  = (__bf16*)((char*)d_ws + (size_t)NT * KSTEPS * 512 * 2); // 6.25 MB

    prep_w<<<(NT * KSTEPS * 512 + 255) / 256, 256, 0, stream>>>(Wx, Wh, Wswz);
    prep_x<<<(B_ * T_ * D_ + 255) / 256, 256, 0, stream>>>(x, xbf);
    for (int t = 0; t < T_; ++t)
        ctlstm_step<<<8, 256, 0, stream>>>(Wswz, xbf, dur, bias, out, t);
}
